// Attention_37203006718300
// MI455X (gfx1250) — matrix operand order repaired and checked by reading, not compiled, not run on hardware
//
#include <hip/hip_runtime.h>
#include <hip/hip_bf16.h>
#include <math.h>

#define BATCH 2
#define SEQ   1024
#define DIM   2048
#define NH    32
#define NKV   8
#define HD    64
#define ROWS  (BATCH * SEQ)   // 2048
#define KVD   (NKV * HD)      // 512

typedef __attribute__((ext_vector_type(16))) _Float16 v16h;
typedef __attribute__((ext_vector_type(8)))  _Float16 v8h;
typedef __attribute__((ext_vector_type(8)))  float    v8f;

// A-fragment (16x32 f16): lane h=lane>>4 holds K = h*16 + (0..15), contiguous.
static __device__ __forceinline__ v16h load_a_frag(const _Float16* p) {
    v8h lo = *(const v8h*)(p);
    v8h hi = *(const v8h*)(p + 8);
    return __builtin_shufflevector(lo, hi, 0, 1, 2, 3, 4, 5, 6, 7,
                                           8, 9, 10, 11, 12, 13, 14, 15);
}
// B-fragment (32x16 f16): lane h holds K = h*16 + (0..15), contiguous.
static __device__ __forceinline__ v16h load_b_frag(const _Float16* p) {
    v8h lo = *(const v8h*)(p);
    v8h hi = *(const v8h*)(p + 8);
    return __builtin_shufflevector(lo, hi, 0, 1, 2, 3, 4, 5, 6, 7,
                                           8, 9, 10, 11, 12, 13, 14, 15);
}

#define WMMA_F16(A, B, C) \
    __builtin_amdgcn_wmma_f32_16x16x32_f16(false, (A), false, (B), (short)0, (C), false, false)

// ---------------------------------------------------------------------------
// WMMA GEMM: Y[M][Nc] = A[M][K] * Bt[Nc][K]^T, fp32 accumulate.
// One wave computes a 32x64 strip: 2 A-frags x 4 B-frags -> 8 WMMAs per
// 32-deep k step (double the B-reuse vs a 16-row strip).
// ---------------------------------------------------------------------------
template <typename OutT>
__global__ __launch_bounds__(256) void gemm16(const _Float16* __restrict__ A,
                                              const _Float16* __restrict__ Bt,
                                              OutT* __restrict__ Y,
                                              int M, int Nc, int K) {
    const int wave = (int)((blockIdx.x * 256u + threadIdx.x) >> 5);
    const int lane = threadIdx.x & 31;
    const int nlo  = lane & 15;
    const int hh   = lane >> 4;
    const int ntiles = Nc >> 6;
    const int mt = wave / ntiles;
    const int nt = wave - mt * ntiles;
    const int m0 = mt << 5;              // 32-row strip
    const int n0 = nt << 6;              // 64-col strip

    const _Float16* arow0 = A  + (size_t)(m0 + nlo) * K + hh * 16;
    const _Float16* arow1 = arow0 + (size_t)16 * K;
    const _Float16* brow0 = Bt + (size_t)(n0 + nlo) * K + hh * 16;
    const size_t bstride = (size_t)16 * K;

    v8f c00 = {}, c01 = {}, c02 = {}, c03 = {};
    v8f c10 = {}, c11 = {}, c12 = {}, c13 = {};
    for (int k = 0; k < K; k += 32) {
        __builtin_prefetch(brow0 + k + 256, 0, 1);
        v16h a0 = load_a_frag(arow0 + k);
        v16h a1 = load_a_frag(arow1 + k);
        v16h b0 = load_b_frag(brow0 + k);
        v16h b1 = load_b_frag(brow0 + bstride + k);
        v16h b2 = load_b_frag(brow0 + 2 * bstride + k);
        v16h b3 = load_b_frag(brow0 + 3 * bstride + k);
        c00 = WMMA_F16(a0, b0, c00);
        c01 = WMMA_F16(a0, b1, c01);
        c02 = WMMA_F16(a0, b2, c02);
        c03 = WMMA_F16(a0, b3, c03);
        c10 = WMMA_F16(a1, b0, c10);
        c11 = WMMA_F16(a1, b1, c11);
        c12 = WMMA_F16(a1, b2, c12);
        c13 = WMMA_F16(a1, b3, c13);
    }
#pragma unroll
    for (int r = 0; r < 8; ++r) {
        const int row = m0 + r + 8 * hh;
        OutT* y0 = Y + (size_t)row * Nc + n0 + nlo;
        y0[0]  = (OutT)c00[r];
        y0[16] = (OutT)c01[r];
        y0[32] = (OutT)c02[r];
        y0[48] = (OutT)c03[r];
        OutT* y1 = y0 + (size_t)16 * Nc;
        y1[0]  = (OutT)c10[r];
        y1[16] = (OutT)c11[r];
        y1[32] = (OutT)c12[r];
        y1[48] = (OutT)c13[r];
    }
}

// ---------------------------------------------------------------------------
// Elementwise helpers
// ---------------------------------------------------------------------------
__global__ void cast16(const float* __restrict__ in, _Float16* __restrict__ out, int n) {
    int i = blockIdx.x * blockDim.x + threadIdx.x;
    if (i < n) out[i] = (_Float16)in[i];
}

// w[K][Nc] (fp32) -> wt[Nc][K] (f16)
__global__ void tcast16(const float* __restrict__ w, _Float16* __restrict__ wt,
                        int K, int Nc) {
    int i = blockIdx.x * blockDim.x + threadIdx.x;
    if (i >= K * Nc) return;
    int k = i / Nc;
    int n = i - k * Nc;
    wt[(size_t)n * K + k] = (_Float16)w[i];
}

// v[b*SEQ+pos][KVD] -> vt[b][KVD][SEQ]
__global__ void vtrans16(const _Float16* __restrict__ v, _Float16* __restrict__ vt) {
    int i = blockIdx.x * blockDim.x + threadIdx.x;
    if (i >= ROWS * KVD) return;
    int row = i / KVD;          // b*SEQ + pos
    int f   = i - row * KVD;
    int b   = row / SEQ;
    int pos = row - b * SEQ;
    vt[(size_t)b * KVD * SEQ + (size_t)f * SEQ + pos] = v[i];
}

// In-place interleaved RoPE on f16 activations laid out [row][nheads*HD].
__global__ void rope16(_Float16* __restrict__ X, int nheads) {
    const int i = blockIdx.x * blockDim.x + threadIdx.x;
    const int total = ROWS * nheads * (HD / 2);
    if (i >= total) return;
    const int pair = i & 31;            // HD/2 = 32
    const int t    = i >> 5;
    const int head = t % nheads;
    const int row  = t / nheads;
    const int pos  = row & (SEQ - 1);
    const float freq = __powf(10000.0f, -(float)pair * (1.0f / 32.0f));
    const float ph = (float)pos * freq;
    float sn, cs;
    __sincosf(ph, &sn, &cs);
    _Float16* p = X + (size_t)row * (nheads * HD) + head * HD + pair * 2;
    const float x0 = (float)p[0];
    const float x1 = (float)p[1];
    p[0] = (_Float16)(x0 * cs - x1 * sn);
    p[1] = (_Float16)(x0 * sn + x1 * cs);
}

// ---------------------------------------------------------------------------
// Flash attention: one wave per (b, head, 16-row q tile). Streams 32-key
// chunks with online softmax; Q*K^T and P*V via WMMA; P staged through LDS
// to convert C-frag layout -> A-frag layout (wave-local, s_wait_dscnt).
// ---------------------------------------------------------------------------
__global__ __launch_bounds__(256) void attn16(const _Float16* __restrict__ Q,
                                              const _Float16* __restrict__ Kc,
                                              const _Float16* __restrict__ Vt,
                                              _Float16* __restrict__ O) {
    __shared__ __align__(16) _Float16 plds[8][16 * 32];
    const int wave = (int)((blockIdx.x * 256u + threadIdx.x) >> 5);
    const int wl   = threadIdx.x >> 5;
    const int lane = threadIdx.x & 31;
    const int nlo  = lane & 15;
    const int hh   = lane >> 4;

    const int qt   = wave & 63;          // SEQ/16 = 64 q tiles
    const int head = (wave >> 6) & 31;
    const int b    = wave >> 11;
    const int j    = head >> 2;          // GQA: rep = 4
    const int q0   = qt << 4;

    // Q A-fragments for the full 64-wide head dim (two 32-deep k steps).
    const _Float16* qbase = Q + (size_t)(b * SEQ + q0 + nlo) * DIM + head * HD + hh * 16;
    const v16h qa0 = load_a_frag(qbase);
    const v16h qa1 = load_a_frag(qbase + 32);

    const _Float16* kbase = Kc + (size_t)(b * SEQ) * KVD + j * HD + hh * 16;
    const _Float16* vb0 = Vt + (size_t)b * KVD * SEQ + (size_t)(j * HD + nlo) * SEQ + hh * 16;
    const _Float16* vb1 = vb0 + (size_t)16 * SEQ;
    const _Float16* vb2 = vb0 + (size_t)32 * SEQ;
    const _Float16* vb3 = vb0 + (size_t)48 * SEQ;

    float mrow[8], lrow[8];
#pragma unroll
    for (int r = 0; r < 8; ++r) { mrow[r] = -INFINITY; lrow[r] = 0.0f; }
    v8f a0 = {}, a1 = {}, a2 = {}, a3 = {};

    _Float16* pl = &plds[wl][0];
    const int ktlast = (q0 + 15) >> 5;
    for (int kt = 0; kt <= ktlast; ++kt) {
        // Scores: 16 q-rows x 32 key-cols, two 16-wide C tiles, K-dim = 64.
        v8f s0 = {}, s1 = {};
        const _Float16* k0 = kbase + (size_t)(kt * 32 + nlo) * KVD;
        const _Float16* k1 = k0 + (size_t)16 * KVD;
        v16h kb;
        kb = load_b_frag(k0);        s0 = WMMA_F16(qa0, kb, s0);
        kb = load_b_frag(k0 + 32);   s0 = WMMA_F16(qa1, kb, s0);
        kb = load_b_frag(k1);        s1 = WMMA_F16(qa0, kb, s1);
        kb = load_b_frag(k1 + 32);   s1 = WMMA_F16(qa1, kb, s1);

#pragma unroll
        for (int r = 0; r < 8; ++r) { s0[r] *= 0.125f; s1[r] *= 0.125f; }

        if (kt == ktlast) {          // causal mask only on the diagonal chunk
#pragma unroll
            for (int r = 0; r < 8; ++r) {
                const int qp = q0 + r + 8 * hh;
                const int kp = (kt << 5) + nlo;
                if (kp > qp)      s0[r] = -INFINITY;
                if (kp + 16 > qp) s1[r] = -INFINITY;
            }
        }

        // Online softmax stats: row reductions across the 16-lane half-groups.
        float alpha[8];
#pragma unroll
        for (int r = 0; r < 8; ++r) {
            float cm = fmaxf(s0[r], s1[r]);
            cm = fmaxf(cm, __shfl_xor(cm, 1));
            cm = fmaxf(cm, __shfl_xor(cm, 2));
            cm = fmaxf(cm, __shfl_xor(cm, 4));
            cm = fmaxf(cm, __shfl_xor(cm, 8));
            const float mnew = fmaxf(mrow[r], cm);
            alpha[r] = __expf(mrow[r] - mnew);
            mrow[r]  = mnew;
        }
#pragma unroll
        for (int r = 0; r < 8; ++r) {
            const float p0 = __expf(s0[r] - mrow[r]);
            const float p1 = __expf(s1[r] - mrow[r]);
            float rs = p0 + p1;
            rs += __shfl_xor(rs, 1);
            rs += __shfl_xor(rs, 2);
            rs += __shfl_xor(rs, 4);
            rs += __shfl_xor(rs, 8);
            lrow[r] = lrow[r] * alpha[r] + rs;
            pl[(r + 8 * hh) * 32 + nlo]      = (_Float16)p0;
            pl[(r + 8 * hh) * 32 + 16 + nlo] = (_Float16)p1;
            a0[r] *= alpha[r];
            a1[r] *= alpha[r];
            a2[r] *= alpha[r];
            a3[r] *= alpha[r];
        }
        asm volatile("s_wait_dscnt 0" ::: "memory");  // wave-local LDS RAW

        // Re-load P as an A-fragment (16 q-rows x 32 key K-dim).
        const v16h pa = load_a_frag(pl + nlo * 32 + hh * 16);
        v16h vf;
        vf = load_b_frag(vb0 + kt * 32);  a0 = WMMA_F16(pa, vf, a0);
        vf = load_b_frag(vb1 + kt * 32);  a1 = WMMA_F16(pa, vf, a1);
        vf = load_b_frag(vb2 + kt * 32);  a2 = WMMA_F16(pa, vf, a2);
        vf = load_b_frag(vb3 + kt * 32);  a3 = WMMA_F16(pa, vf, a3);
    }

#pragma unroll
    for (int r = 0; r < 8; ++r) {
        const float inv = 1.0f / lrow[r];
        _Float16* o = O + (size_t)(b * SEQ + q0 + r + 8 * hh) * DIM + head * HD + nlo;
        o[0]  = (_Float16)(a0[r] * inv);
        o[16] = (_Float16)(a1[r] * inv);
        o[32] = (_Float16)(a2[r] * inv);
        o[48] = (_Float16)(a3[r] * inv);
    }
}

// ---------------------------------------------------------------------------
extern "C" void kernel_launch(void* const* d_in, const int* in_sizes, int n_in,
                              void* d_out, int out_size, void* d_ws, size_t ws_size,
                              hipStream_t stream) {
    const float* x  = (const float*)d_in[0];
    const float* wq = (const float*)d_in[1];
    const float* wk = (const float*)d_in[2];
    const float* wv = (const float*)d_in[3];
    const float* wo = (const float*)d_in[4];
    float* out = (float*)d_out;

    char* ws = (char*)d_ws;
    size_t off = 0;
    auto alloc = [&](size_t bytes) -> char* {
        char* p = ws + off;
        off += (bytes + 255) & ~(size_t)255;
        return p;
    };
    _Float16* xh  = (_Float16*)alloc((size_t)ROWS * DIM * 2);
    _Float16* wqt = (_Float16*)alloc((size_t)DIM * DIM * 2);
    _Float16* wkt = (_Float16*)alloc((size_t)KVD * DIM * 2);
    _Float16* wvt = (_Float16*)alloc((size_t)KVD * DIM * 2);
    _Float16* wot = (_Float16*)alloc((size_t)DIM * DIM * 2);
    _Float16* qh  = (_Float16*)alloc((size_t)ROWS * DIM * 2);
    _Float16* kh  = (_Float16*)alloc((size_t)ROWS * KVD * 2);
    _Float16* vh  = (_Float16*)alloc((size_t)ROWS * KVD * 2);
    _Float16* vt  = (_Float16*)alloc((size_t)ROWS * KVD * 2);
    _Float16* ah  = (_Float16*)alloc((size_t)ROWS * DIM * 2);

    const dim3 blk(256);
    cast16<<<(ROWS * DIM + 255) / 256, blk, 0, stream>>>(x, xh, ROWS * DIM);
    tcast16<<<(DIM * DIM + 255) / 256, blk, 0, stream>>>(wq, wqt, DIM, DIM);
    tcast16<<<(DIM * KVD + 255) / 256, blk, 0, stream>>>(wk, wkt, DIM, KVD);
    tcast16<<<(DIM * KVD + 255) / 256, blk, 0, stream>>>(wv, wvt, DIM, KVD);
    tcast16<<<(DIM * DIM + 255) / 256, blk, 0, stream>>>(wo, wot, DIM, DIM);

    // 32x64 strips: waves = (M/32) * (Nc/64), 8 waves per block.
    gemm16<_Float16><<<(ROWS / 32) * (DIM / 64) / 8, blk, 0, stream>>>(xh, wqt, qh, ROWS, DIM, DIM);
    gemm16<_Float16><<<(ROWS / 32) * (KVD / 64) / 8, blk, 0, stream>>>(xh, wkt, kh, ROWS, KVD, DIM);
    gemm16<_Float16><<<(ROWS / 32) * (KVD / 64) / 8, blk, 0, stream>>>(xh, wvt, vh, ROWS, KVD, DIM);

    rope16<<<(ROWS * NH  * 32 + 255) / 256, blk, 0, stream>>>(qh, NH);
    rope16<<<(ROWS * NKV * 32 + 255) / 256, blk, 0, stream>>>(kh, NKV);
    vtrans16<<<(ROWS * KVD + 255) / 256, blk, 0, stream>>>(vh, vt);

    attn16<<<(BATCH * NH * (SEQ / 16)) * 32 / 256, blk, 0, stream>>>(qh, kh, vt, ah);

    gemm16<float><<<(ROWS / 32) * (DIM / 64) / 8, blk, 0, stream>>>(ah, wot, out, ROWS, DIM, DIM);
}
